// LSR_84284438217063
// MI455X (gfx1250) — matrix operand order repaired and checked by reading, not compiled, not run on hardware
//
#include <hip/hip_runtime.h>
#include <hip/hip_bf16.h>

// ---------------------------------------------------------------------------
// MI455X (gfx1250) implementation. All heavy matmuls use
// V_WMMA_F32_16X16X32_BF16 (bf16 in, f32 accumulate).
//
// Dims: B=4 L=1024 H=256 NE=64 P=512 E=65536 R=4, N=256, D2=512,
//       DIN=140(->160), DBI=276(->288), Kbil=288*288=82944, REL=97(->112)
// ---------------------------------------------------------------------------

typedef __attribute__((ext_vector_type(16))) __bf16 v16bf;
typedef __attribute__((ext_vector_type(8)))  float  v8f;

#define LL     1024
#define HH     256
#define G4     1024          // 4*H
#define NN     256           // B*NE
#define D2     512
#define DINP   160           // padded DIN (140)
#define NPAIR  2048          // B*P
#define DBIP   288           // padded DBI (276)
#define KBIL   (288*288)     // 82944
#define KBC    (KBIL/32)     // 2592

// -------------------------- WMMA tile helpers ------------------------------
// A tile: 16x32 bf16, row-major source, lda in elements (multiple of 8).
// Lane l: row m=l&15, khalf=l>>4; a[0..7]=A[m][kh*8..], a[8..15]=A[m][16+kh*8..]
__device__ __forceinline__ v16bf load_a_tile(const __bf16* A, int lda) {
  int lane = threadIdx.x & 31;
  int m = lane & 15, kh = lane >> 4;
  const __bf16* p0 = A + (size_t)m * lda + kh * 8;
  union { v16bf v; uint4 q[2]; } u;
  u.q[0] = *(const uint4*)(p0);
  u.q[1] = *(const uint4*)(p0 + 16);
  return u.v;
}

// B tile: 32x16 bf16 (KxN) stored pre-transposed N-major: Bt[n][k], ldb = K.
// Lane l: col n=l&15, khalf=l>>4; b[0..7]=Bt[n][kh*8..], b[8..15]=Bt[n][16+kh*8..]
__device__ __forceinline__ v16bf load_b_tileT(const __bf16* Bt, int ldb) {
  int lane = threadIdx.x & 31;
  int n = lane & 15, kh = lane >> 4;
  const __bf16* p = Bt + (size_t)n * ldb + kh * 8;
  union { v16bf v; uint4 q[2]; } u;
  u.q[0] = *(const uint4*)(p);
  u.q[1] = *(const uint4*)(p + 16);
  return u.v;
}

__device__ __forceinline__ v8f wmma_bf(v16bf a, v16bf b, v8f c) {
  return __builtin_amdgcn_wmma_f32_16x16x32_bf16(
      /*neg_a=*/false, a, /*neg_b=*/false, b,
      /*c_mod=*/(short)0, c, /*reuse_a=*/false, /*reuse_b=*/false);
}

__device__ __forceinline__ float sigm(float x) {
  return 1.0f / (1.0f + __expf(-x));
}

// -------------------------- prep kernels -----------------------------------
// dst[n][k] (bf16, [Ndst][Kpad]) = src[k][n] (f32 [Ksrc][Nsrc]) with zero pad
__global__ void convert_T(const float* src, __bf16* dst,
                          int Ksrc, int Nsrc, int Kpad, int Ndst) {
  int idx = blockIdx.x * blockDim.x + threadIdx.x;
  if (idx >= Ndst * Kpad) return;
  int n = idx / Kpad, k = idx % Kpad;
  float v = (k < Ksrc && n < Nsrc) ? src[(size_t)k * Nsrc + n] : 0.0f;
  dst[idx] = (__bf16)v;
}

__global__ void cvt_f2bf(const float* s, __bf16* d, int n) {
  int i = blockIdx.x * blockDim.x + threadIdx.x;
  if (i < n) d[i] = (__bf16)s[i];
}

__global__ void cvt_i2bf(const int* s, __bf16* d, int n) {
  int i = blockIdx.x * blockDim.x + threadIdx.x;
  if (i < n) d[i] = (__bf16)(float)s[i];
}

__global__ void zero_i32(int* p, int n) {
  int i = blockIdx.x * blockDim.x + threadIdx.x;
  if (i < n) p[i] = 0;
}

// X[l*4+b][c] = concat(word, coref, ner) embeddings, bf16, zero-padded to 160
__global__ void build_X(const int* ctx, const int* pos, const int* ner,
                        const float* wt, const float* ct, const float* nt,
                        __bf16* X) {
  int idx = blockIdx.x * blockDim.x + threadIdx.x;
  if (idx >= 4096 * DINP) return;
  int row = idx / DINP, c = idx % DINP;
  int l = row >> 2, b = row & 3;
  float v = 0.0f;
  if (c < 100)      v = wt[(size_t)ctx[b * LL + l] * 100 + c];
  else if (c < 120) v = ct[(size_t)pos[b * LL + l] * 20 + (c - 100)];
  else if (c < 140) v = nt[(size_t)ner[b * LL + l] * 20 + (c - 120)];
  X[idx] = (__bf16)v;
}

// Wt[k][i*288+j] = bili_W[i][j][k] (bf16, zero pad i,j>=276)
__global__ void conv_bili(const float* W, __bf16* Wt) {
  long idx = (long)blockIdx.x * blockDim.x + threadIdx.x;
  if (idx >= (long)HH * KBIL) return;
  int k = (int)(idx / KBIL);
  int rem = (int)(idx % KBIL);
  int i = rem / DBIP, j = rem % DBIP;
  float v = (i < 276 && j < 276) ? W[((long)(i * 276 + j)) * HH + k] : 0.0f;
  Wt[idx] = (__bf16)v;
}

// C_r[dst][src] edge counts (turns segment_sum into a GEMM)
__global__ void edge_count(const int* esrc, const int* edst, const int* etyp,
                           int* C) {
  int e = blockIdx.x * blockDim.x + threadIdx.x;
  if (e < 65536)
    atomicAdd(&C[((size_t)etyp[e] * NN + edst[e]) * NN + esrc[e]], 1);
}

// ------------------------ generic WMMA GEMM --------------------------------
// D = relu( sum_{rep}(A0_rep @ B0_repT) [+ A1 @ B1T] [+ bias] )
// A row-major bf16 [M x K]; Bt N-major bf16 [N x K]. One 16x16 tile per wave.
__global__ void gemm_bf16(const __bf16* A0, const __bf16* B0,
                          const __bf16* A1, const __bf16* B1,
                          int nrep, int Asz, int Bsz,
                          const float* bias, float* Cf, __bf16* Crm, __bf16* Ct,
                          int M, int Npad, int K, int ldo, int relu,
                          long Ab, long Bb, long Cfb, long Crmb) {
  int wave = threadIdx.x >> 5;
  int lane = threadIdx.x & 31;
  int Ntiles = Npad >> 4;
  int ntile = blockIdx.x * 8 + wave;
  if (ntile >= Ntiles) return;
  int mtile = blockIdx.y;
  int z = blockIdx.z;
  int Kt = K >> 5;
  v8f acc = {};
  for (int rep = 0; rep < nrep; ++rep) {
    const __bf16* Ap = A0 + (size_t)z * Ab + (size_t)rep * Asz +
                       (size_t)mtile * 16 * K;
    const __bf16* Bp = B0 + (size_t)z * Bb + (size_t)rep * Bsz +
                       (size_t)ntile * 16 * K;
    for (int kt = 0; kt < Kt; ++kt) {
      v16bf a = load_a_tile(Ap + kt * 32, K);
      v16bf b = load_b_tileT(Bp + kt * 32, K);
      acc = wmma_bf(a, b, acc);
    }
  }
  if (A1 != nullptr) {
    const __bf16* Ap = A1 + (size_t)mtile * 16 * K;
    const __bf16* Bp = B1 + (size_t)ntile * 16 * K;
    for (int kt = 0; kt < Kt; ++kt) {
      v16bf a = load_a_tile(Ap + kt * 32, K);
      v16bf b = load_b_tileT(Bp + kt * 32, K);
      acc = wmma_bf(a, b, acc);
    }
  }
  int n = lane & 15, kh = lane >> 4;
  int col = ntile * 16 + n;
  int row0 = mtile * 16 + 8 * kh;
  if (bias) {
    float bv = bias[col];
#pragma unroll
    for (int v = 0; v < 8; ++v) acc[v] += bv;
  }
  if (relu) {
#pragma unroll
    for (int v = 0; v < 8; ++v) acc[v] = fmaxf(acc[v], 0.0f);
  }
  if (Cf) {
#pragma unroll
    for (int v = 0; v < 8; ++v)
      Cf[(size_t)z * Cfb + (size_t)(row0 + v) * Npad + col] = acc[v];
  }
  if (Crm) {
#pragma unroll
    for (int v = 0; v < 8; ++v)
      Crm[(size_t)z * Crmb + (size_t)(row0 + v) * ldo + col] = (__bf16)acc[v];
  }
  if (Ct) {  // store transposed bf16: Ct[col][row], ld = M
#pragma unroll
    for (int v = 0; v < 8; ++v)
      Ct[(size_t)col * M + row0 + v] = (__bf16)acc[v];
  }
}

// ----------------------------- BiLSTM --------------------------------------
// Input projection X@Wi precomputed (Xg, f32, bias folded in). This kernel
// runs only the sequential recurrence: per step G[4x1024] = Hpad16 @ WhT via
// WMMA (batch padded to 16), gate nonlinearity in VALU, h kept in LDS bf16.
// One workgroup (8 waves) per direction; writes docsT[b][d][l] (transposed,
// N-major, bf16) directly as the B-operand for the entity-pooling GEMM.
__global__ void lstm_kernel(const float* Xg_f, const float* Xg_b,
                            const __bf16* WhT_f, const __bf16* WhT_b,
                            __bf16* docsT) {
  int dir = blockIdx.x;
  const float* Xg = dir ? Xg_b : Xg_f;
  const __bf16* WhT = dir ? WhT_b : WhT_f;
  __shared__ __align__(16) __bf16 hA[16 * HH];   // padded-batch hidden state
  __shared__ float G[4 * G4];                    // gates for real batches
  int tid = threadIdx.x;  // block = 256 (8 waves)
  for (int c = tid; c < 16 * HH; c += 256) hA[c] = (__bf16)0.0f;
  float cst[4] = {0.0f, 0.0f, 0.0f, 0.0f};
  __syncthreads();
  int wave = tid >> 5, lane = tid & 31;
  int n = lane & 15, kh = lane >> 4;
  for (int t = 0; t < LL; ++t) {
    int l = dir ? (LL - 1 - t) : t;
    v16bf areg[8];
#pragma unroll
    for (int kt = 0; kt < 8; ++kt)
      areg[kt] = load_a_tile((const __bf16*)hA + kt * 32, HH);
    for (int ni = 0; ni < 8; ++ni) {
      int ntile = wave * 8 + ni;
      v8f acc = {};
#pragma unroll
      for (int kt = 0; kt < 8; ++kt) {
        v16bf b = load_b_tileT(WhT + (size_t)ntile * 16 * HH + kt * 32, HH);
        acc = wmma_bf(areg[kt], b, acc);
      }
      int col = ntile * 16 + n;
      if (kh == 0) {  // rows 0..3 = real batches
#pragma unroll
        for (int v = 0; v < 4; ++v)
          G[v * G4 + col] = acc[v] + Xg[(size_t)(l * 4 + v) * G4 + col];
      }
    }
    __syncthreads();
    int j = tid;  // hidden unit 0..255
#pragma unroll
    for (int b = 0; b < 4; ++b) {
      float gi = sigm(G[b * G4 + j]);
      float gf = sigm(G[b * G4 + 256 + j]);
      float gg = tanhf(G[b * G4 + 512 + j]);
      float go = sigm(G[b * G4 + 768 + j]);
      float c = gf * cst[b] + gi * gg;
      cst[b] = c;
      float h = go * tanhf(c);
      hA[b * HH + j] = (__bf16)h;
      docsT[((size_t)b * D2 + dir * HH + j) * LL + l] = (__bf16)h;
    }
    __syncthreads();
  }
}

// ------------------------- pair gather / fill ------------------------------
__global__ void pair_gather(const int* hts, const __bf16* hs, const __bf16* ts,
                            __bf16* hp, __bf16* tp) {
  int row = blockIdx.x;        // 0..2047
  int t = threadIdx.x;         // 64 threads, uint4 = 8 bf16 each
  int b = row >> 9, p = row & 511;
  int hi = b * 64 + hts[(b * 512 + p) * 2 + 0];
  int ti = b * 64 + hts[(b * 512 + p) * 2 + 1];
  ((uint4*)(hp + (size_t)row * D2))[t] = ((const uint4*)(hs + (size_t)hi * D2))[t];
  ((uint4*)(tp + (size_t)row * D2))[t] = ((const uint4*)(ts + (size_t)ti * D2))[t];
}

// pad hd/td cols 256..287 with distance embeddings (+zeros), and write
// feat[:,256..511] = hd*td (cols 0..255)
__global__ void fill_pairs(const float* dis_table, const int* dht,
                           const int* dth, __bf16* hd, __bf16* td,
                           __bf16* feat) {
  int row = blockIdx.x, t = threadIdx.x;  // block = 320
  if (t < 256) {
    float a = (float)hd[(size_t)row * DBIP + t];
    float b = (float)td[(size_t)row * DBIP + t];
    feat[(size_t)row * D2 + 256 + t] = (__bf16)(a * b);
  } else if (t < 288) {
    int q = t - 256;
    float v = (q < 20) ? dis_table[(size_t)dht[row] * 20 + q] : 0.0f;
    hd[(size_t)row * DBIP + 256 + q] = (__bf16)v;
  } else {
    int q = t - 288;
    float v = (q < 20) ? dis_table[(size_t)dth[row] * 20 + q] : 0.0f;
    td[(size_t)row * DBIP + 256 + q] = (__bf16)v;
  }
}

// --------------------------- bilinear (80 GFLOP) ---------------------------
// bl[row,k] = sum_{i,j} hd_i W[i,j,k] td_j  ==  (hd (x) td)[row,:] @ Wt[k,:]
// A built on the fly: A[row, i*288+j] = hd_i * td_j (LDS-staged hd/td).
// Wt [256 x 82944] bf16 is L2-resident (42.5 MB < 192 MB L2).
__global__ void bilinear_kernel(const __bf16* hd, const __bf16* td,
                                const __bf16* Wt, const float* bili_b,
                                __bf16* feat) {
  __shared__ float hdsh[16 * DBIP];
  __shared__ float tdsh[16 * DBIP];
  int m0 = blockIdx.x * 16;
  for (int c = threadIdx.x; c < 16 * DBIP; c += 256) {  // 4608 = 18*256
    int r = c / DBIP, q = c % DBIP;
    hdsh[c] = (float)hd[(size_t)(m0 + r) * DBIP + q];
    tdsh[c] = (float)td[(size_t)(m0 + r) * DBIP + q];
  }
  __syncthreads();
  int wave = threadIdx.x >> 5, lane = threadIdx.x & 31;
  int m = lane & 15, kh = lane >> 4;
  int nt0 = wave * 2;  // each wave owns 2 of 16 N-tiles
  const __bf16* W0 = Wt + (size_t)(nt0 * 16) * KBIL;
  const __bf16* W1 = Wt + (size_t)((nt0 + 1) * 16) * KBIL;
  v8f acc0 = {}, acc1 = {};
  int i = 0, j0 = 0;
  for (int kc = 0; kc < KBC; ++kc) {
    float hv = hdsh[m * DBIP + i];
    v16bf a;
#pragma unroll
    for (int q = 0; q < 8; ++q) {
      a[q]     = (__bf16)(hv * tdsh[m * DBIP + j0 + kh * 8 + q]);
      a[q + 8] = (__bf16)(hv * tdsh[m * DBIP + j0 + 16 + kh * 8 + q]);
    }
    size_t f0 = (size_t)kc * 32;
    __builtin_prefetch(W0 + f0 + 2048, 0, 1);
    __builtin_prefetch(W1 + f0 + 2048, 0, 1);
    v16bf b0 = load_b_tileT(W0 + f0, KBIL);
    v16bf b1 = load_b_tileT(W1 + f0, KBIL);
    acc0 = wmma_bf(a, b0, acc0);
    acc1 = wmma_bf(a, b1, acc1);
    j0 += 32;
    if (j0 == DBIP) { j0 = 0; ++i; }
  }
  int nn = lane & 15;
#pragma unroll
  for (int s = 0; s < 2; ++s) {
    v8f acc = s ? acc1 : acc0;
    int col = (nt0 + s) * 16 + nn;
    float bb = bili_b[col];
#pragma unroll
    for (int v = 0; v < 8; ++v) {
      float val = fmaxf(acc[v] + bb, 0.0f);
      feat[(size_t)(m0 + v + 8 * kh) * D2 + col] = (__bf16)val;
    }
  }
}

// ------------------------- final projection --------------------------------
// logits = feat[2048x512] @ out_W[512x97] + out_b  (N padded 97->112, 7 waves)
__global__ void out_gemm(const __bf16* A, const __bf16* Bt, const float* ob,
                         float* out) {
  int wave = threadIdx.x >> 5;  // block = 224 -> 7 waves = 7 N-tiles
  int lane = threadIdx.x & 31;
  int mtile = blockIdx.x;
  v8f acc = {};
  for (int kt = 0; kt < 16; ++kt) {
    v16bf a = load_a_tile(A + (size_t)mtile * 16 * D2 + kt * 32, D2);
    v16bf b = load_b_tileT(Bt + (size_t)wave * 16 * D2 + kt * 32, D2);
    acc = wmma_bf(a, b, acc);
  }
  int n = lane & 15, kh = lane >> 4;
  int col = wave * 16 + n;
  if (col < 97) {
    float bb = ob[col];
#pragma unroll
    for (int v = 0; v < 8; ++v)
      out[(size_t)(mtile * 16 + v + 8 * kh) * 97 + col] = acc[v] + bb;
  }
}

// ============================ host orchestration ===========================
static inline void gemm_go(hipStream_t s, const __bf16* A0, const __bf16* B0,
                           const __bf16* A1, const __bf16* B1, int nrep,
                           int Asz, int Bsz, const float* bias, float* Cf,
                           __bf16* Crm, __bf16* Ct, int M, int N, int K,
                           int ldo, int relu, long Ab, long Bb, long Cfb,
                           long Crmb, int batch) {
  dim3 g((N / 16 + 7) / 8, M / 16, batch);
  gemm_bf16<<<g, 256, 0, s>>>(A0, B0, A1, B1, nrep, Asz, Bsz, bias, Cf, Crm,
                              Ct, M, N, K, ldo, relu, Ab, Bb, Cfb, Crmb);
}

extern "C" void kernel_launch(void* const* d_in, const int* in_sizes, int n_in,
                              void* d_out, int out_size, void* d_ws,
                              size_t ws_size, hipStream_t stream) {
  const float* e_mapping = (const float*)d_in[0];
  const float* word_t    = (const float*)d_in[1];
  const float* coref_t   = (const float*)d_in[2];
  const float* ner_t     = (const float*)d_in[3];
  const float* Wi_f = (const float*)d_in[4];
  const float* Wh_f = (const float*)d_in[5];
  const float* b_f  = (const float*)d_in[6];
  const float* Wi_b = (const float*)d_in[7];
  const float* Wh_b = (const float*)d_in[8];
  const float* b_b  = (const float*)d_in[9];
  const float* W_rel    = (const float*)d_in[10];
  const float* W_self_h = (const float*)d_in[11];
  const float* W_msg_h  = (const float*)d_in[12];
  const float* b_h      = (const float*)d_in[13];
  const float* W_self_t = (const float*)d_in[14];
  const float* W_msg_t  = (const float*)d_in[15];
  const float* b_t      = (const float*)d_in[16];
  const float* head_W = (const float*)d_in[17];
  const float* head_b = (const float*)d_in[18];
  const float* tail_W = (const float*)d_in[19];
  const float* tail_b = (const float*)d_in[20];
  const float* dis_table = (const float*)d_in[21];
  const float* bili_W = (const float*)d_in[22];
  const float* bili_b = (const float*)d_in[23];
  const float* out_W  = (const float*)d_in[24];
  const float* out_b  = (const float*)d_in[25];
  const int* ctx_idx = (const int*)d_in[26];
  const int* pos_idx = (const int*)d_in[27];
  const int* ner_idx = (const int*)d_in[28];
  const int* hts     = (const int*)d_in[29];
  const int* dht     = (const int*)d_in[30];
  const int* dth     = (const int*)d_in[31];
  const int* esrc    = (const int*)d_in[32];
  const int* edst    = (const int*)d_in[33];
  const int* etyp    = (const int*)d_in[34];
  float* out = (float*)d_out;

  // -------- workspace carve (~102 MB) --------
  char* w = (char*)d_ws;
  auto carve = [&](size_t bytes) {
    void* p = (void*)w;
    w += (bytes + 255) & ~(size_t)255;
    return p;
  };
  __bf16* X_bf   = (__bf16*)carve((size_t)4096 * DINP * 2);
  __bf16* WiTf   = (__bf16*)carve((size_t)G4 * DINP * 2);
  __bf16* WiTb   = (__bf16*)carve((size_t)G4 * DINP * 2);
  __bf16* WhTf   = (__bf16*)carve((size_t)G4 * HH * 2);
  __bf16* WhTb   = (__bf16*)carve((size_t)G4 * HH * 2);
  float*  Xgf    = (float*)carve((size_t)4096 * G4 * 4);
  float*  Xgb    = (float*)carve((size_t)4096 * G4 * 4);
  __bf16* docsT  = (__bf16*)carve((size_t)4 * D2 * LL * 2);
  __bf16* emapbf = (__bf16*)carve((size_t)4 * 64 * LL * 2);
  __bf16* ent_bf = (__bf16*)carve((size_t)NN * D2 * 2);
  __bf16* WrelT  = (__bf16*)carve((size_t)4 * D2 * D2 * 2);
  __bf16* mT     = (__bf16*)carve((size_t)4 * D2 * NN * 2);
  int*    Cnt    = (int*)carve((size_t)4 * NN * NN * 4);
  __bf16* Cbf    = (__bf16*)carve((size_t)4 * NN * NN * 2);
  __bf16* agg_bf = (__bf16*)carve((size_t)NN * D2 * 2);
  __bf16* WshT   = (__bf16*)carve((size_t)D2 * D2 * 2);
  __bf16* WmhT   = (__bf16*)carve((size_t)D2 * D2 * 2);
  __bf16* WstT   = (__bf16*)carve((size_t)D2 * D2 * 2);
  __bf16* WmtT   = (__bf16*)carve((size_t)D2 * D2 * 2);
  __bf16* hWT    = (__bf16*)carve((size_t)HH * D2 * 2);
  __bf16* tWT    = (__bf16*)carve((size_t)HH * D2 * 2);
  __bf16* hs_bf  = (__bf16*)carve((size_t)NN * D2 * 2);
  __bf16* ts_bf  = (__bf16*)carve((size_t)NN * D2 * 2);
  __bf16* hp_bf  = (__bf16*)carve((size_t)NPAIR * D2 * 2);
  __bf16* tp_bf  = (__bf16*)carve((size_t)NPAIR * D2 * 2);
  __bf16* hd_bf  = (__bf16*)carve((size_t)NPAIR * DBIP * 2);
  __bf16* td_bf  = (__bf16*)carve((size_t)NPAIR * DBIP * 2);
  __bf16* featbf = (__bf16*)carve((size_t)NPAIR * D2 * 2);
  __bf16* WbT    = (__bf16*)carve((size_t)HH * KBIL * 2);
  __bf16* oWT    = (__bf16*)carve((size_t)112 * D2 * 2);
  (void)ws_size; (void)n_in; (void)in_sizes; (void)out_size;

  auto g1 = [](int n) { return dim3((n + 255) / 256); };

  // 1) embeddings -> X (bf16, padded), weight transposes
  build_X<<<g1(4096 * DINP), 256, 0, stream>>>(ctx_idx, pos_idx, ner_idx,
                                               word_t, coref_t, ner_t, X_bf);
  convert_T<<<g1(G4 * DINP), 256, 0, stream>>>(Wi_f, WiTf, 140, G4, DINP, G4);
  convert_T<<<g1(G4 * DINP), 256, 0, stream>>>(Wi_b, WiTb, 140, G4, DINP, G4);
  convert_T<<<g1(G4 * HH), 256, 0, stream>>>(Wh_f, WhTf, HH, G4, HH, G4);
  convert_T<<<g1(G4 * HH), 256, 0, stream>>>(Wh_b, WhTb, HH, G4, HH, G4);

  // 2) input projection GEMMs (bias folded), then sequential LSTM recurrence
  gemm_go(stream, X_bf, WiTf, nullptr, nullptr, 1, 0, 0, b_f, Xgf, nullptr,
          nullptr, 4096, G4, DINP, 0, 0, 0, 0, 0, 0, 1);
  gemm_go(stream, X_bf, WiTb, nullptr, nullptr, 1, 0, 0, b_b, Xgb, nullptr,
          nullptr, 4096, G4, DINP, 0, 0, 0, 0, 0, 0, 1);
  lstm_kernel<<<2, 256, 0, stream>>>(Xgf, Xgb, WhTf, WhTb, docsT);

  // 3) entity pooling: ent[b] = e_map[b][64x1024] @ docs[b][1024x512]
  cvt_f2bf<<<g1(4 * 64 * LL), 256, 0, stream>>>(e_mapping, emapbf, 4 * 64 * LL);
  gemm_go(stream, emapbf, docsT, nullptr, nullptr, 1, 0, 0, nullptr, nullptr,
          ent_bf, nullptr, 64, D2, LL, D2, 0, (long)64 * LL, (long)D2 * LL, 0,
          (long)64 * D2, 4);

  // 4) per-relation messages m_r = relu(ent @ W_rel[r]), stored transposed
  for (int r = 0; r < 4; ++r) {
    convert_T<<<g1(D2 * D2), 256, 0, stream>>>(W_rel + (size_t)r * D2 * D2,
                                               WrelT + (size_t)r * D2 * D2,
                                               D2, D2, D2, D2);
    gemm_go(stream, ent_bf, WrelT + (size_t)r * D2 * D2, nullptr, nullptr, 1,
            0, 0, nullptr, nullptr, nullptr, mT + (size_t)r * D2 * NN, NN, D2,
            D2, 0, 1, 0, 0, 0, 0, 1);
  }

  // 5) scatter-add -> count-matrix GEMM: agg = sum_r C_r @ m_r
  zero_i32<<<g1(4 * NN * NN), 256, 0, stream>>>(Cnt, 4 * NN * NN);
  edge_count<<<256, 256, 0, stream>>>(esrc, edst, etyp, Cnt);
  cvt_i2bf<<<g1(4 * NN * NN), 256, 0, stream>>>(Cnt, Cbf, 4 * NN * NN);
  gemm_go(stream, Cbf, mT, nullptr, nullptr, 4, NN * NN, D2 * NN, nullptr,
          nullptr, agg_bf, nullptr, NN, D2, NN, D2, 0, 0, 0, 0, 0, 1);

  // 6) hs/ts = relu(ent@W_self + agg@W_msg + b)
  convert_T<<<g1(D2 * D2), 256, 0, stream>>>(W_self_h, WshT, D2, D2, D2, D2);
  convert_T<<<g1(D2 * D2), 256, 0, stream>>>(W_msg_h,  WmhT, D2, D2, D2, D2);
  convert_T<<<g1(D2 * D2), 256, 0, stream>>>(W_self_t, WstT, D2, D2, D2, D2);
  convert_T<<<g1(D2 * D2), 256, 0, stream>>>(W_msg_t,  WmtT, D2, D2, D2, D2);
  gemm_go(stream, ent_bf, WshT, agg_bf, WmhT, 1, 0, 0, b_h, nullptr, hs_bf,
          nullptr, NN, D2, D2, D2, 1, 0, 0, 0, 0, 1);
  gemm_go(stream, ent_bf, WstT, agg_bf, WmtT, 1, 0, 0, b_t, nullptr, ts_bf,
          nullptr, NN, D2, D2, D2, 1, 0, 0, 0, 0, 1);

  // 7) gather pairs, project heads/tails (into hd/td, ldo=288)
  pair_gather<<<NPAIR, 64, 0, stream>>>(hts, hs_bf, ts_bf, hp_bf, tp_bf);
  convert_T<<<g1(HH * D2), 256, 0, stream>>>(head_W, hWT, D2, HH, D2, HH);
  convert_T<<<g1(HH * D2), 256, 0, stream>>>(tail_W, tWT, D2, HH, D2, HH);
  gemm_go(stream, hp_bf, hWT, nullptr, nullptr, 1, 0, 0, head_b, nullptr,
          hd_bf, nullptr, NPAIR, HH, D2, DBIP, 0, 0, 0, 0, 0, 1);
  gemm_go(stream, tp_bf, tWT, nullptr, nullptr, 1, 0, 0, tail_b, nullptr,
          td_bf, nullptr, NPAIR, HH, D2, DBIP, 0, 0, 0, 0, 0, 1);
  fill_pairs<<<NPAIR, 320, 0, stream>>>(dis_table, dht, dth, hd_bf, td_bf,
                                        featbf);

  // 8) bilinear scoring (dominant 80 GFLOP) + final projection
  conv_bili<<<g1(HH * KBIL), 256, 0, stream>>>(bili_W, WbT);
  bilinear_kernel<<<NPAIR / 16, 256, 0, stream>>>(hd_bf, td_bf, WbT, bili_b,
                                                  featbf);
  convert_T<<<g1(112 * D2), 256, 0, stream>>>(out_W, oWT, D2, 97, D2, 112);
  out_gemm<<<NPAIR / 16, 224, 0, stream>>>(featbf, oWT, out_b, out);
}
